// QuantizerWrapper_13039520710805
// MI455X (gfx1250) — matrix operand order repaired and checked by reading, not compiled, not run on hardware
//
#include <hip/hip_runtime.h>

typedef __attribute__((ext_vector_type(16))) __bf16 v16bf;
typedef __attribute__((ext_vector_type(4)))  __bf16 v4bf;
typedef __attribute__((ext_vector_type(8)))  float  v8f;

#define B_TOK 32768
#define DIM   512
#define KCB   8192
#define CT    32                  // codebook columns per tile
#define NTILE (KCB / CT)          // 256
#define ROWS_PER_BLOCK 128

// LDS tile geometry: Bs[2][32][264] bf16 (single shared object -> LDS offset 0)
#define ROW_BYTES 528              // 264 bf16
#define BUF_BYTES (32 * ROW_BYTES) // 16896

// ---------------- fp32 -> bf16 codebook conversion ----------------
__global__ void cvt_bf16_kernel(const float* __restrict__ in,
                                __bf16* __restrict__ out, int n4) {
  int i = blockIdx.x * blockDim.x + threadIdx.x;
  if (i >= n4) return;
  const float4 f = ((const float4*)in)[i];
  v4bf o = { (__bf16)f.x, (__bf16)f.y, (__bf16)f.z, (__bf16)f.w };
  *(v4bf*)(out + (size_t)i * 4) = o;
}

// ---------------- per-code squared norm ----------------
__global__ void e2_kernel(const float* __restrict__ e, float* __restrict__ e2) {
  __shared__ float s[8];
  const int row = blockIdx.x;
  const int t = threadIdx.x;
  const float a0 = e[(size_t)row * DIM + t];
  const float a1 = e[(size_t)row * DIM + 256 + t];
  float v = a0 * a0 + a1 * a1;
#pragma unroll
  for (int m = 16; m >= 1; m >>= 1) v += __shfl_xor(v, m, 32);
  if ((t & 31) == 0) s[t >> 5] = v;
  __syncthreads();
  if (t < 8) {
    float w = s[t];
#pragma unroll
    for (int m = 4; m >= 1; m >>= 1) w += __shfl_xor(w, m, 32);
    if (t == 0) e2[row] = w;
  }
}

// ---------------- WMMA helpers ----------------
__device__ __forceinline__ v8f wmma_bf16(v16bf a, v16bf b, v8f c) {
  return __builtin_amdgcn_wmma_f32_16x16x32_bf16(false, a, false, b,
                                                 (short)0, c, false, false);
}

__device__ __forceinline__ v16bf lds_frag(const __bf16* p) {
  union { uint4 q[2]; v16bf v; } u;
  u.q[0] = *(const uint4*)(p);
  u.q[1] = *(const uint4*)(p + 8);
  return u.v;
}

// ---------------- main: scores + running argmax ----------------
__global__ __launch_bounds__(256) void vq_argmin_kernel(
    const float* __restrict__ x, const __bf16* __restrict__ eb,
    const float* __restrict__ e2, int* __restrict__ ind) {
  // double-buffered B tile: 32 codes x 256 dims (bf16), +16B row pad
  __shared__ __bf16 Bs[2][CT][264];   // 33,792 bytes, LDS offset 0

  const int t    = threadIdx.x;
  const int lane = t & 31;
  const int wave = t >> 5;
  const int n    = lane & 15;
  const int hi   = lane >> 4;
  const int rowBase = blockIdx.x * ROWS_PER_BLOCK + wave * 16;

  // ---- load this wave's 16x512 A block once, fp32->bf16, into 16 fragments
  v16bf a[16];
  {
    const float* xr = x + (size_t)(rowBase + n) * DIM;
#pragma unroll
    for (int f = 0; f < 16; ++f) {
      const int d0 = f * 32 + hi * 16;
      float4 q0 = *(const float4*)(xr + d0);
      float4 q1 = *(const float4*)(xr + d0 + 4);
      float4 q2 = *(const float4*)(xr + d0 + 8);
      float4 q3 = *(const float4*)(xr + d0 + 12);
      a[f][0]  = (__bf16)q0.x;  a[f][1]  = (__bf16)q0.y;
      a[f][2]  = (__bf16)q0.z;  a[f][3]  = (__bf16)q0.w;
      a[f][4]  = (__bf16)q1.x;  a[f][5]  = (__bf16)q1.y;
      a[f][6]  = (__bf16)q1.z;  a[f][7]  = (__bf16)q1.w;
      a[f][8]  = (__bf16)q2.x;  a[f][9]  = (__bf16)q2.y;
      a[f][10] = (__bf16)q2.z;  a[f][11] = (__bf16)q2.w;
      a[f][12] = (__bf16)q3.x;  a[f][13] = (__bf16)q3.y;
      a[f][14] = (__bf16)q3.z;  a[f][15] = (__bf16)q3.w;
    }
  }

  float bestv[8];
  int   besti[8];
#pragma unroll
  for (int r = 0; r < 8; ++r) { bestv[r] = -3.402823466e38f; besti[r] = 0; }

  // ---- async global->LDS fill (no VGPR staging, tracked by ASYNCcnt).
  // IOFFSET adds to BOTH the global and the LDS address, so one base covers
  // this thread's 64 contiguous bytes with offset:0/16/32/48.
  // IMPORTANT: &Bs[...] is passed as an (unused) asm operand so the LDS array
  // escapes -- otherwise GlobalOpt proves "no stores to Bs" and folds every
  // ds_load of it to poison (observed in round 2).
  const int fr  = t >> 3;   // 0..31 : code row within tile
  const int seg = t & 7;    // 0..7  : 32-bf16 segment
  auto async_fill = [&](int buf, int step) {
    const int col0  = (step >> 1) * CT;
    const int phase = step & 1;
    const __bf16* gp =
        eb + (size_t)(col0 + fr) * DIM + phase * 256 + seg * 32;
    const __bf16* lp = &Bs[buf][fr][seg * 32];   // escapes into the asm
    unsigned lds = (unsigned)(buf * BUF_BYTES + fr * ROW_BYTES + seg * 64);
    asm volatile(
        "global_load_async_to_lds_b128 %0, %1, off\n\t"
        "global_load_async_to_lds_b128 %0, %1, off offset:16\n\t"
        "global_load_async_to_lds_b128 %0, %1, off offset:32\n\t"
        "global_load_async_to_lds_b128 %0, %1, off offset:48"
        :: "v"(lds), "v"(gp), "v"(lp) : "memory");
  };

  async_fill(0, 0);
  asm volatile("s_wait_asynccnt 0" ::: "memory");
  __syncthreads();

  for (int tile = 0; tile < NTILE; ++tile) {
    const int col0 = tile * CT;
    v8f c0 = {0.f, 0.f, 0.f, 0.f, 0.f, 0.f, 0.f, 0.f};
    v8f c1 = c0;

    // issue next-buffer fill + e2 loads early; compute covers their latency
    async_fill(1, 2 * tile + 1);
    const float e2v0 = e2[col0 + n];
    const float e2v1 = e2[col0 + 16 + n];

    // ---- phase 0: buf0 = dims [0,256); preload fragment lf+1 before wmma(lf)
    {
      v16bf b0 = lds_frag(&Bs[0][n][hi * 16]);
      v16bf b1 = lds_frag(&Bs[0][16 + n][hi * 16]);
#pragma unroll
      for (int lf = 0; lf < 8; ++lf) {
        v16bf nb0 = b0, nb1 = b1;
        if (lf < 7) {
          nb0 = lds_frag(&Bs[0][n][(lf + 1) * 32 + hi * 16]);
          nb1 = lds_frag(&Bs[0][16 + n][(lf + 1) * 32 + hi * 16]);
        }
        c0 = wmma_bf16(a[lf], b0, c0);
        c1 = wmma_bf16(a[lf], b1, c1);
        b0 = nb0; b1 = nb1;
      }
    }
    asm volatile("s_wait_asynccnt 0" ::: "memory");
    __syncthreads();

    // ---- phase 1: buf1 = dims [256,512)
    if (tile + 1 < NTILE) async_fill(0, 2 * tile + 2);
    {
      v16bf b0 = lds_frag(&Bs[1][n][hi * 16]);
      v16bf b1 = lds_frag(&Bs[1][16 + n][hi * 16]);
#pragma unroll
      for (int lf = 0; lf < 8; ++lf) {
        v16bf nb0 = b0, nb1 = b1;
        if (lf < 7) {
          nb0 = lds_frag(&Bs[1][n][(lf + 1) * 32 + hi * 16]);
          nb1 = lds_frag(&Bs[1][16 + n][(lf + 1) * 32 + hi * 16]);
        }
        c0 = wmma_bf16(a[8 + lf], b0, c0);
        c1 = wmma_bf16(a[8 + lf], b1, c1);
        b0 = nb0; b1 = nb1;
      }
    }

    // ---- running argmax of (2*x.e - ||e||^2); ascending cols => first-max ties
#pragma unroll
    for (int r = 0; r < 8; ++r) {
      const float v0 = 2.0f * c0[r] - e2v0;
      if (v0 > bestv[r]) { bestv[r] = v0; besti[r] = col0 + n; }
      const float v1 = 2.0f * c1[r] - e2v1;
      if (v1 > bestv[r]) { bestv[r] = v1; besti[r] = col0 + 16 + n; }
    }
    asm volatile("s_wait_asynccnt 0" ::: "memory");
    __syncthreads();
  }

  // ---- cross-lane argmax within each 16-lane half (xor masks stay in-half)
#pragma unroll
  for (int r = 0; r < 8; ++r) {
    float v = bestv[r];
    int   i = besti[r];
#pragma unroll
    for (int m = 1; m <= 8; m <<= 1) {
      const float ov = __shfl_xor(v, m, 32);
      const int   oi = __shfl_xor(i, m, 32);
      if (ov > v || (ov == v && oi < i)) { v = ov; i = oi; }
    }
    if (n == 0) ind[rowBase + r + 8 * hi] = i;
  }
}

// ---------------- gather: quantize = embed[ind], emit indices ----------------
__global__ void gather_kernel(const float* __restrict__ embed,
                              const int* __restrict__ ind,
                              float* __restrict__ outq,
                              int* __restrict__ outi) {
  const int row = blockIdx.x;
  const int t = threadIdx.x;
  const int k = ind[row];
  const float4* src = (const float4*)(embed + (size_t)k * DIM);
  float4* dst = (float4*)(outq + (size_t)row * DIM);
  dst[t] = src[t];
  if (t == 0) outi[row] = k;
}

extern "C" void kernel_launch(void* const* d_in, const int* in_sizes, int n_in,
                              void* d_out, int out_size, void* d_ws, size_t ws_size,
                              hipStream_t stream) {
  (void)in_sizes; (void)n_in; (void)out_size; (void)ws_size;
  const float* x     = (const float*)d_in[0];
  const float* embed = (const float*)d_in[1];

  char* w = (char*)d_ws;
  __bf16* eb = (__bf16*)w;                                   // 8 MB bf16 codebook
  const size_t eb_bytes = (size_t)KCB * DIM * sizeof(__bf16);
  float* e2 = (float*)(w + eb_bytes);                        // 32 KB norms
  int*   ind = (int*)(w + eb_bytes + (size_t)KCB * sizeof(float)); // 128 KB indices

  float* outq = (float*)d_out;
  int*   outi = (int*)((float*)d_out + (size_t)B_TOK * DIM);

  const int n4 = (KCB * DIM) / 4;
  cvt_bf16_kernel<<<(n4 + 255) / 256, 256, 0, stream>>>(embed, eb, n4);
  e2_kernel<<<KCB, 256, 0, stream>>>(embed, e2);
  vq_argmin_kernel<<<B_TOK / ROWS_PER_BLOCK, 256, 0, stream>>>(x, eb, e2, ind);
  gather_kernel<<<B_TOK, 128, 0, stream>>>(embed, ind, outq, outi);
}
